// MultiBoxLoss_24352464568755
// MI455X (gfx1250) — compile-verified, matrix operand order checked by reading, not run on hardware
//
#include <hip/hip_runtime.h>
#include <cstddef>
#include <cstdint>

// ---------------------------------------------------------------------------
// MultiBoxLoss for MI455X (gfx1250, wave32).
// Hard-negative-mining rank selection is done with V_WMMA_I32_16X16X64_IU8:
// rank(i) = #{j: m[j]>m[i]} + #{j<i: m[j]==m[i]} computed as a 0/1-byte
// comparison matrix times an all-ones byte matrix on the matrix pipe.
// ---------------------------------------------------------------------------

#define BATCH  128
#define NPRI   8732
#define NCLS   21
#define NOBJ   16
#define PPAD   8768                 // multiple of 64, >= NPRI, = 137 * 64
#define NTILES ((NPRI + 15) / 16)   // 546 i-tiles of 16 rows
#define MSPLIT 4                    // blocks per batch row in mb_mine
#define NWORDS (PPAD / 32)          // 274 pos-mask words

typedef int v8i_t __attribute__((ext_vector_type(8)));

__device__ __forceinline__ float smooth_l1(float d) {
  float a = fabsf(d);
  return (a < 1.0f) ? 0.5f * a * a : a - 0.5f;
}

// ---------------------------------------------------------------------------
// K0: zero accumulators and per-truth argmax keys
// ---------------------------------------------------------------------------
__global__ void mb_init(unsigned long long* __restrict__ keys,
                        int* __restrict__ num_pos,
                        float* __restrict__ acc_f,   // [0]=loss_l, [1]=loss_c
                        int* __restrict__ acc_i) {   // [0]=total_pos
  int t = threadIdx.x;
  for (int i = t; i < BATCH * NOBJ; i += blockDim.x) keys[i] = 0ull;
  for (int i = t; i < BATCH; i += blockDim.x) num_pos[i] = 0;
  if (t == 0) { acc_f[0] = 0.0f; acc_f[1] = 0.0f; acc_i[0] = 0; }
}

// ---------------------------------------------------------------------------
// K1: per-prior best truth (IoU argmax over 16 truths, first-max wins) and
//     per-truth best prior via packed u64 key max: (iou_bits<<32)|~p.
//     IoU >= 0 so float bits are monotone; ~p breaks ties toward smaller p
//     (== jnp.argmax first-occurrence).  LDS ds_max_u64 pre-reduction.
// ---------------------------------------------------------------------------
__global__ void mb_match1(const float* __restrict__ priors,
                          const float* __restrict__ truths,
                          float* __restrict__ bto,   // [B][P] best truth overlap
                          int*   __restrict__ bti,   // [B][P] best truth idx
                          unsigned long long* __restrict__ keys) {  // [B][NO]
  __shared__ float4 t_sh[NOBJ];
  __shared__ unsigned long long ksh[NOBJ];
  const int b = blockIdx.y;
  const int p = blockIdx.x * blockDim.x + threadIdx.x;

  if (threadIdx.x < NOBJ) {
    t_sh[threadIdx.x] = ((const float4*)truths)[b * NOBJ + threadIdx.x];
    ksh[threadIdx.x]  = 0ull;
  }
  __syncthreads();

  if (p < NPRI) {
    float4 pr  = ((const float4*)priors)[p];
    float px0 = pr.x - 0.5f * pr.z, py0 = pr.y - 0.5f * pr.w;
    float px1 = pr.x + 0.5f * pr.z, py1 = pr.y + 0.5f * pr.w;
    float ap  = pr.z * pr.w;
    float best = -1.0f; int bidx = 0;
    #pragma unroll
    for (int t = 0; t < NOBJ; ++t) {
      float4 tb = t_sh[t];
      float ix = fmaxf(fminf(tb.z, px1) - fmaxf(tb.x, px0), 0.0f);
      float iy = fmaxf(fminf(tb.w, py1) - fmaxf(tb.y, py0), 0.0f);
      float inter = ix * iy;
      float at = (tb.z - tb.x) * (tb.w - tb.y);
      float iou = inter / (at + ap - inter);
      if (iou > best) { best = iou; bidx = t; }   // strict > : first max kept
      unsigned long long key =
          ((unsigned long long)__float_as_uint(iou) << 32) |
          (unsigned int)(~(unsigned int)p);
      atomicMax(&ksh[t], key);                    // ds_max_u64
    }
    bto[(size_t)b * NPRI + p] = best;
    bti[(size_t)b * NPRI + p] = bidx;
  }
  __syncthreads();
  if (threadIdx.x < NOBJ)
    atomicMax(&keys[b * NOBJ + threadIdx.x], ksh[threadIdx.x]);
}

// ---------------------------------------------------------------------------
// K2: forced matches, conf targets, box encoding, smooth-L1 over positives
// ---------------------------------------------------------------------------
__global__ void mb_match2(const float* __restrict__ loc,
                          const float* __restrict__ priors,
                          const float* __restrict__ truths,
                          const int*   __restrict__ labels,
                          const float* __restrict__ bto,
                          const int*   __restrict__ bti,
                          const unsigned long long* __restrict__ keys,
                          int*   __restrict__ conf_t,
                          int*   __restrict__ num_pos,
                          int*   __restrict__ total_pos,
                          float* __restrict__ loss_l_acc) {
  __shared__ int    bp[NOBJ];
  __shared__ float4 t_sh[NOBJ];
  __shared__ int    lab_sh[NOBJ];
  __shared__ float  redf[256];
  __shared__ int    redi[256];
  const int b   = blockIdx.y;
  const int p   = blockIdx.x * blockDim.x + threadIdx.x;
  const int tid = threadIdx.x;

  if (tid < NOBJ) {
    bp[tid]     = (int)(~(unsigned int)(keys[b * NOBJ + tid] & 0xFFFFFFFFull));
    t_sh[tid]   = ((const float4*)truths)[b * NOBJ + tid];
    lab_sh[tid] = labels[b * NOBJ + tid];
  }
  __syncthreads();

  float l1 = 0.0f; int pc = 0;
  if (p < NPRI) {
    int forced = -1;
    #pragma unroll
    for (int t = 0; t < NOBJ; ++t)
      if (bp[t] == p) forced = t;                 // ascending t: max t wins
    size_t idx = (size_t)b * NPRI + p;
    float ov = (forced >= 0) ? 2.0f : bto[idx];
    int   ti = (forced >= 0) ? forced : bti[idx];
    int  cls = (ov < 0.5f) ? 0 : lab_sh[ti];
    conf_t[idx] = cls;
    if (cls > 0) {
      pc = 1;
      float4 mt = t_sh[ti];
      float4 pr = ((const float4*)priors)[p];
      float gx = ((mt.x + mt.z) * 0.5f - pr.x) / (0.1f * pr.z);
      float gy = ((mt.y + mt.w) * 0.5f - pr.y) / (0.1f * pr.w);
      float gw = logf((mt.z - mt.x) / pr.z) / 0.2f;
      float gh = logf((mt.w - mt.y) / pr.w) / 0.2f;
      float4 ld = ((const float4*)loc)[idx];
      l1 = smooth_l1(ld.x - gx) + smooth_l1(ld.y - gy) +
           smooth_l1(ld.z - gw) + smooth_l1(ld.w - gh);
    }
  }
  redf[tid] = l1; redi[tid] = pc;
  __syncthreads();
  for (int st = 128; st > 0; st >>= 1) {
    if (tid < st) { redf[tid] += redf[tid + st]; redi[tid] += redi[tid + st]; }
    __syncthreads();
  }
  if (tid == 0) {
    atomicAdd(loss_l_acc, redf[0]);
    atomicAdd(&num_pos[b], redi[0]);
    atomicAdd(total_pos, redi[0]);
  }
}

// ---------------------------------------------------------------------------
// K3: fused score pass.  ce = lse - x[gt] equals the mining score, so one
//     read of conf_data (94 MB) covers both losses.
// ---------------------------------------------------------------------------
__global__ void mb_score(const float* __restrict__ conf,
                         const int*   __restrict__ conf_t,
                         float* __restrict__ s_out,
                         float* __restrict__ m_out) {
  size_t idx = (size_t)blockIdx.x * blockDim.x + threadIdx.x;
  if (idx >= (size_t)BATCH * NPRI) return;
  const float* x = conf + idx * NCLS;
  float mx = x[0];
  #pragma unroll
  for (int c = 1; c < NCLS; ++c) mx = fmaxf(mx, x[c]);
  float se = 0.0f;
  #pragma unroll
  for (int c = 0; c < NCLS; ++c) se += expf(x[c] - mx);
  float lse = logf(se) + mx;
  int gt = conf_t[idx];
  float sc = lse - x[gt];        // >= 0 always (lse >= max >= x[gt])
  s_out[idx] = sc;
  m_out[idx] = (gt > 0) ? 0.0f : sc;
}

// ---------------------------------------------------------------------------
// K4: hard-negative mining via IU8 WMMA rank counting.
//     Grid {BATCH, MSPLIT}: each block stages the full m-row in LDS and owns
//     a strided subset of i-tiles (full j-range per tile -> exact ranks).
//     A[i][k] = (m[j]>m[i]) | (m[j]==m[i] & j<i) as bytes, B = ones bytes.
//     Positive mask is balloted into an LDS bitmask so the inner loop never
//     touches conf_t; s is fetched from global only for selected rows.
// ---------------------------------------------------------------------------
__global__ void mb_mine(const float* __restrict__ s,
                        const float* __restrict__ m,
                        const int*   __restrict__ conf_t,
                        const int*   __restrict__ num_pos,
                        float* __restrict__ loss_c_acc) {
  __shared__ float    mlds[PPAD];
  __shared__ unsigned posmask[NWORDS];
  __shared__ float    red[256];
  const int b     = blockIdx.x;
  const int split = blockIdx.y;
  const int tid   = threadIdx.x;
  const int lane  = tid & 31;
  const int wave  = tid >> 5;                 // 8 wave32s per block
  const float* mrow = m + (size_t)b * NPRI;
  const int* ctrow  = conf_t + (size_t)b * NPRI;

  for (int i = tid; i < PPAD; i += blockDim.x)
    mlds[i] = (i < NPRI) ? mrow[i] : -1.0f;   // pad < all real m (m >= 0)

  // ballot conf_t>0 into a bitmask: one word per 32 priors
  for (int w = wave; w < NWORDS; w += 8) {
    int i = w * 32 + lane;
    bool isp = (i < NPRI) && (ctrow[i] > 0);
    unsigned long long bal = __ballot(isp);   // wave32: low 32 bits
    if (lane == 0) posmask[w] = (unsigned)bal;
  }
  __syncthreads();

  const int nneg = min(3 * num_pos[b], NPRI - 1);
  const int M    = lane & 15;
  const int half = lane >> 4;

  const v8i_t ones = {0x01010101, 0x01010101, 0x01010101, 0x01010101,
                      0x01010101, 0x01010101, 0x01010101, 0x01010101};
  float lsum = 0.0f;

  // 32 wave-streams per batch row across MSPLIT blocks
  for (int tile = split * 8 + wave; tile < NTILES; tile += 8 * MSPLIT) {
    const int ibase = tile * 16;
    const int i     = ibase + M;              // < PPAD always (8736 <= 8768)
    const float mi  = mlds[i];
    v8i_t acc = {0, 0, 0, 0, 0, 0, 0, 0};

    for (int chunk = 0; chunk < PPAD; chunk += 64) {
      v8i_t a;
      #pragma unroll
      for (int v = 0; v < 8; ++v) {
        // 8-bit A 16x64 VGPR layout (ISA 7.12.2): K base per (vgpr, lane-half)
        int kb = ((v & 1) << 2) | (((v >> 1) & 1) << 4) |
                 (((v >> 2) & 1) << 5) | (half << 3);
        int j0 = chunk + kb;
        float4 mj = *reinterpret_cast<const float4*>(&mlds[j0]); // ds_load_b128
        unsigned b0 = ((mj.x > mi) | ((mj.x == mi) & ((j0 + 0) < i))) ? 1u : 0u;
        unsigned b1 = ((mj.y > mi) | ((mj.y == mi) & ((j0 + 1) < i))) ? 1u : 0u;
        unsigned b2 = ((mj.z > mi) | ((mj.z == mi) & ((j0 + 2) < i))) ? 1u : 0u;
        unsigned b3 = ((mj.w > mi) | ((mj.w == mi) & ((j0 + 3) < i))) ? 1u : 0u;
        a[v] = (int)(b0 | (b1 << 8) | (b2 << 16) | (b3 << 24));
      }
      // v_wmma_i32_16x16x64_iu8 : D = A(0/1 bytes) x ones + C  -> rank counts
      acc = __builtin_amdgcn_wmma_i32_16x16x64_iu8(false, a, false, ones, acc,
                                                   false, false);
    }

    // D layout: VGPR v, lanes 0-15 -> M=v, lanes 16-31 -> M=v+8; all N equal.
    if ((lane & 15) == 0) {
      #pragma unroll
      for (int v = 0; v < 8; ++v) {
        int ii = ibase + v + (half << 3);
        if (ii < NPRI) {
          bool isp = (posmask[ii >> 5] >> (ii & 31)) & 1u;
          bool isn = acc[v] < nneg;
          if (isp | isn) lsum += s[(size_t)b * NPRI + ii];
        }
      }
    }
  }

  red[tid] = lsum;
  __syncthreads();
  for (int st = 128; st > 0; st >>= 1) {
    if (tid < st) red[tid] += red[tid + st];
    __syncthreads();
  }
  if (tid == 0) atomicAdd(loss_c_acc, red[0]);
}

// ---------------------------------------------------------------------------
// K5: finalize
// ---------------------------------------------------------------------------
__global__ void mb_final(const float* __restrict__ acc_f,
                         const int*   __restrict__ acc_i,
                         float* __restrict__ out) {
  if (threadIdx.x == 0) {
    float N = fmaxf((float)acc_i[0], 1.0f);
    out[0] = acc_f[0] / N;
    out[1] = acc_f[1] / N;
  }
}

// ---------------------------------------------------------------------------
extern "C" void kernel_launch(void* const* d_in, const int* in_sizes, int n_in,
                              void* d_out, int out_size, void* d_ws,
                              size_t ws_size, hipStream_t stream) {
  const float* loc    = (const float*)d_in[0];
  const float* conf   = (const float*)d_in[1];
  const float* priors = (const float*)d_in[2];
  const float* truths = (const float*)d_in[3];
  const int*   labels = (const int*)d_in[4];
  float* out = (float*)d_out;

  char* ws = (char*)d_ws;
  const size_t BP = (size_t)BATCH * NPRI;
  size_t o_keys = 0;                                           // 16 KB
  size_t o_npos = o_keys + (size_t)BATCH * NOBJ * 8;           // 512 B
  size_t o_acc  = o_npos + 512;                                // 64 B
  size_t o_bto  = o_acc + 64;
  size_t o_bti  = o_bto + BP * 4;
  size_t o_ct   = o_bti + BP * 4;
  size_t o_s    = o_ct  + BP * 4;
  size_t o_m    = o_s   + BP * 4;

  unsigned long long* keys = (unsigned long long*)(ws + o_keys);
  int*   npos = (int*)(ws + o_npos);
  float* accf = (float*)(ws + o_acc);         // [0]=loss_l, [1]=loss_c
  int*   acci = (int*)(ws + o_acc + 8);       // [0]=total_pos
  float* bto  = (float*)(ws + o_bto);
  int*   bti  = (int*)(ws + o_bti);
  int*   ct   = (int*)(ws + o_ct);
  float* sarr = (float*)(ws + o_s);
  float* marr = (float*)(ws + o_m);

  mb_init<<<1, 256, 0, stream>>>(keys, npos, accf, acci);

  dim3 gmatch((NPRI + 255) / 256, BATCH);
  mb_match1<<<gmatch, 256, 0, stream>>>(priors, truths, bto, bti, keys);
  mb_match2<<<gmatch, 256, 0, stream>>>(loc, priors, truths, labels, bto, bti,
                                        keys, ct, npos, acci, &accf[0]);

  int gscore = (int)((BP + 255) / 256);
  mb_score<<<gscore, 256, 0, stream>>>(conf, ct, sarr, marr);

  dim3 gmine(BATCH, MSPLIT);
  mb_mine<<<gmine, 256, 0, stream>>>(sarr, marr, ct, npos, &accf[1]);

  mb_final<<<1, 32, 0, stream>>>(accf, acci, out);
}
